// Decoder_80487687127096
// MI455X (gfx1250) — matrix operand order repaired and checked by reading, not compiled, not run on hardware
//
#include <hip/hip_runtime.h>
#include <hip/hip_bf16.h>

// ---------------- problem constants (from reference) ----------------
#define VOCAB 16000
#define EMB   512
#define HID   1024
#define T_LEN 64
#define BATCH 64
#define PAD_IDX 0
#define SOS_IDX 1
#define G3H (3 * HID)

typedef __attribute__((ext_vector_type(16))) __bf16 v16bf;
typedef __attribute__((ext_vector_type(8)))  __bf16 v8bf;
typedef __attribute__((ext_vector_type(8)))  float  v8f;

// ---------------- helpers ----------------
__device__ __forceinline__ unsigned short f32_to_bf16_bits(float f) {
    union { float f; unsigned u; } v; v.f = f;
    unsigned u = v.u;
    if ((u & 0x7F800000u) == 0x7F800000u) {        // Inf / NaN: keep payload
        return (unsigned short)(u >> 16);
    }
    unsigned r = u + 0x7FFFu + ((u >> 16) & 1u);   // round-to-nearest-even
    return (unsigned short)(r >> 16);
}

__device__ __forceinline__ float sigmoidf_(float x) {
    return 1.0f / (1.0f + expf(-x));
}

// ---------------- one-time conversion: f32 -> bf16 ----------------
__global__ void dec_cvt_bf16(const float* __restrict__ src,
                             unsigned short* __restrict__ dst, int n) {
    int stride = gridDim.x * blockDim.x;
    for (int i = blockIdx.x * blockDim.x + threadIdx.x; i < n; i += stride)
        dst[i] = f32_to_bf16_bits(src[i]);
}

// ---------------- init: h = encoder_hidden[0], inp = SOS ----------------
__global__ void dec_init(const float* __restrict__ enc,
                         float* __restrict__ h,
                         unsigned short* __restrict__ h_bf,
                         int* __restrict__ inp) {
    int i = blockIdx.x * blockDim.x + threadIdx.x;
    if (i < BATCH * HID) {
        float v = enc[i];                // encoder_hidden is [1,B,H]
        h[i] = v;
        h_bf[i] = f32_to_bf16_bits(v);
    }
    if (i < BATCH) inp[i] = SOS_IDX;
}

// ---------------- per-step: gather embedding rows -> bf16 ----------------
__global__ void dec_gather_x(const float* __restrict__ emb,
                             const int* __restrict__ inp,
                             unsigned short* __restrict__ x_bf) {
    int i = blockIdx.x * blockDim.x + threadIdx.x;
    if (i >= BATCH * EMB) return;
    int b = i / EMB, e = i - b * EMB;
    int row = inp[b];
    float v = (row == PAD_IDX) ? 0.0f : emb[(size_t)row * EMB + e];
    x_bf[i] = f32_to_bf16_bits(v);
}

// ---------------- bf16 WMMA GEMM: out[64,N] = A[64,K] @ W[N,K]^T + bias ----
// grid = (N/64), block = 128 (4 waves); wave w owns N-tile blockIdx.x*4+w.
// Each wave computes ALL FOUR M-tiles (M=64) with 4 independent accumulators:
//   - the streamed W fragment (the bandwidth-dominant operand) is loaded ONCE
//     per K-step and reused 4x, cutting L2 weight traffic 4x;
//   - 4 independent WMMA accumulation chains issue back-to-back into the
//     matrix pipe with no C-chaining stalls between them.
__global__ void __launch_bounds__(128)
dec_gemm_bf16(const unsigned short* __restrict__ A,   // [64 x K] row-major
              const unsigned short* __restrict__ W,   // [N  x K] row-major
              const float* __restrict__ bias,         // [N]
              float* __restrict__ out,                // [64 x N]
              int N, int K) {
    const int wave = threadIdx.x >> 5;
    const int lane = threadIdx.x & 31;
    const int half = lane >> 4;          // 0: lanes 0-15, 1: lanes 16-31
    const int l15  = lane & 15;
    const int n_tile = blockIdx.x * 4 + wave;
    const int n0 = n_tile * 16;

    // A: 16x32 tile in contiguous k order (k = 16*half + s, not the ISA table order) — m = lane&15 (+16 per m-tile); K halves split at 16
    const unsigned short* abase = A + (size_t)l15 * K;
    // B: 32x16 tile in contiguous k order (k = 16*half + s, same as A) — n = lane&15; K halves split at 16
    const unsigned short* wrow = W + (size_t)(n0 + l15) * K + half * 16;

    v8f acc0 = {}, acc1 = {}, acc2 = {}, acc3 = {};
    const size_t mstride = (size_t)16 * K;            // elements between m-tiles

    for (int k0 = 0; k0 < K; k0 += 32) {
        __builtin_prefetch((const void*)(wrow + k0 + 256), 0, 0); // stream W
        v8bf b0 = *reinterpret_cast<const v8bf*>(wrow + k0);
        v8bf b1 = *reinterpret_cast<const v8bf*>(wrow + k0 + 8);
        v16bf bfm = __builtin_shufflevector(b0, b1, 0,1,2,3,4,5,6,7,
                                                     8,9,10,11,12,13,14,15);
        const unsigned short* ak = abase + k0 + half * 16;
#pragma unroll
        for (int mt = 0; mt < 4; ++mt) {
            v8bf a0 = *reinterpret_cast<const v8bf*>(ak + mt * mstride);
            v8bf a1 = *reinterpret_cast<const v8bf*>(ak + mt * mstride + 8);
            v16bf af = __builtin_shufflevector(a0, a1, 0,1,2,3,4,5,6,7,
                                                        8,9,10,11,12,13,14,15);
            v8f c = (mt == 0) ? acc0 : (mt == 1) ? acc1 : (mt == 2) ? acc2 : acc3;
            c = __builtin_amdgcn_wmma_f32_16x16x32_bf16(
                    false, af, false, bfm, (short)0, c, false, false);
            if      (mt == 0) acc0 = c;
            else if (mt == 1) acc1 = c;
            else if (mt == 2) acc2 = c;
            else              acc3 = c;
        }
    }

    const float bn = bias[n0 + l15];
    // C/D layout: element r -> row M = mt*16 + r + 8*half, col N = lane&15
#pragma unroll
    for (int mt = 0; mt < 4; ++mt) {
        v8f c = (mt == 0) ? acc0 : (mt == 1) ? acc1 : (mt == 2) ? acc2 : acc3;
        float* orow = out + (size_t)(mt * 16 + half * 8) * N + n0 + l15;
#pragma unroll
        for (int r = 0; r < 8; ++r)
            orow[(size_t)r * N] = c[r] + bn;
    }
}

// ---------------- fused GRU pointwise update ----------------
__global__ void dec_gru_update(const float* __restrict__ gx,   // [B,3H]
                               const float* __restrict__ gh,   // [B,3H]
                               float* __restrict__ h,          // [B,H] in/out
                               unsigned short* __restrict__ h_bf) {
    int i = blockIdx.x * blockDim.x + threadIdx.x;
    if (i >= BATCH * HID) return;
    int b = i >> 10, j = i & (HID - 1);
    const float* gxr = gx + (size_t)b * G3H;
    const float* ghr = gh + (size_t)b * G3H;
    float r = sigmoidf_(gxr[j] + ghr[j]);
    float z = sigmoidf_(gxr[HID + j] + ghr[HID + j]);
    float n = tanhf(gxr[2 * HID + j] + r * ghr[2 * HID + j]);
    float hn = (1.0f - z) * n + z * h[i];
    h[i] = hn;
    h_bf[i] = f32_to_bf16_bits(hn);
}

// ---------------- log_softmax + first-occurrence argmax ----------------
// one block per batch row; out_t points at d_out + t*V (row stride T*V)
__global__ void __launch_bounds__(256)
dec_logsoftmax(const float* __restrict__ logits,  // [B,V]
               float* __restrict__ out_t,
               int* __restrict__ inp_next) {
    __shared__ float smax[256];
    __shared__ int   sidx[256];
    __shared__ float ssum[256];
    const int b = blockIdx.x, tid = threadIdx.x;
    const float* row = logits + (size_t)b * VOCAB;

    float m = -3.402823466e38f; int mi = VOCAB;
    for (int v = tid; v < VOCAB; v += 256) {
        float x = row[v];
        if (x > m) { m = x; mi = v; }    // keeps first occurrence in-thread
    }
    smax[tid] = m; sidx[tid] = mi;
    __syncthreads();
    for (int s = 128; s > 0; s >>= 1) {
        if (tid < s) {
            float a = smax[tid], c = smax[tid + s];
            if (c > a || (c == a && sidx[tid + s] < sidx[tid])) {
                smax[tid] = c; sidx[tid] = sidx[tid + s];
            }
        }
        __syncthreads();
    }
    const float rowmax = smax[0];

    float sum = 0.0f;
    for (int v = tid; v < VOCAB; v += 256) sum += expf(row[v] - rowmax);
    ssum[tid] = sum;
    __syncthreads();
    for (int s = 128; s > 0; s >>= 1) {
        if (tid < s) ssum[tid] += ssum[tid + s];
        __syncthreads();
    }
    const float logZ = rowmax + logf(ssum[0]);

    float* orow = out_t + (size_t)b * T_LEN * VOCAB;  // [B,T,V], t folded in
    for (int v = tid; v < VOCAB; v += 256) orow[v] = row[v] - logZ;
    if (tid == 0) inp_next[b] = sidx[0];
}

// ---------------- tail: decoder_hidden -> end of d_out ----------------
__global__ void dec_copy_hidden(const float* __restrict__ h,
                                float* __restrict__ out_tail) {
    int i = blockIdx.x * blockDim.x + threadIdx.x;
    if (i < BATCH * HID) out_tail[i] = h[i];
}

// ==================== host-side launch ====================
extern "C" void kernel_launch(void* const* d_in, const int* in_sizes, int n_in,
                              void* d_out, int out_size, void* d_ws, size_t ws_size,
                              hipStream_t stream) {
    (void)in_sizes; (void)n_in; (void)out_size; (void)ws_size;
    // input order: target(0) enc_hidden(1) emb(2) w_ih(3) w_hh(4) b_ih(5) b_hh(6) fc_w(7) fc_b(8)
    const float* enc  = (const float*)d_in[1];
    const float* emb  = (const float*)d_in[2];
    const float* w_ih = (const float*)d_in[3];
    const float* w_hh = (const float*)d_in[4];
    const float* b_ih = (const float*)d_in[5];
    const float* b_hh = (const float*)d_in[6];
    const float* fc_w = (const float*)d_in[7];
    const float* fc_b = (const float*)d_in[8];
    float* out = (float*)d_out;

    // ---- carve workspace (all offsets 256B aligned) ----
    char* ws = (char*)d_ws;
    size_t off = 0;
    auto carve = [&](size_t bytes) -> void* {
        void* p = ws + off;
        off += (bytes + 255) & ~(size_t)255;
        return p;
    };
    unsigned short* w_ih_bf = (unsigned short*)carve((size_t)G3H * EMB * 2);
    unsigned short* w_hh_bf = (unsigned short*)carve((size_t)G3H * HID * 2);
    unsigned short* fc_w_bf = (unsigned short*)carve((size_t)VOCAB * HID * 2);
    unsigned short* x_bf    = (unsigned short*)carve((size_t)BATCH * EMB * 2);
    unsigned short* h_bf    = (unsigned short*)carve((size_t)BATCH * HID * 2);
    float* h      = (float*)carve((size_t)BATCH * HID * 4);
    float* gx     = (float*)carve((size_t)BATCH * G3H * 4);
    float* gh     = (float*)carve((size_t)BATCH * G3H * 4);
    float* logits = (float*)carve((size_t)BATCH * VOCAB * 4);
    int*   inp    = (int*)carve((size_t)BATCH * 4);

    // ---- one-time weight conversion (deterministic every call) ----
    dec_cvt_bf16<<<2048, 256, 0, stream>>>(w_ih, w_ih_bf, G3H * EMB);
    dec_cvt_bf16<<<4096, 256, 0, stream>>>(w_hh, w_hh_bf, G3H * HID);
    dec_cvt_bf16<<<8192, 256, 0, stream>>>(fc_w, fc_w_bf, VOCAB * HID);
    dec_init<<<(BATCH * HID + 255) / 256, 256, 0, stream>>>(enc, h, h_bf, inp);

    // ---- sequential decode loop ----
    for (int t = 0; t < T_LEN; ++t) {
        dec_gather_x<<<(BATCH * EMB + 255) / 256, 256, 0, stream>>>(emb, inp, x_bf);
        // gx = x @ w_ih^T + b_ih   (M=64, N=3072, K=512)
        dec_gemm_bf16<<<G3H / 64, 128, 0, stream>>>(
            x_bf, w_ih_bf, b_ih, gx, G3H, EMB);
        // gh = h @ w_hh^T + b_hh   (M=64, N=3072, K=1024)
        dec_gemm_bf16<<<G3H / 64, 128, 0, stream>>>(
            h_bf, w_hh_bf, b_hh, gh, G3H, HID);
        dec_gru_update<<<(BATCH * HID + 255) / 256, 256, 0, stream>>>(gx, gh, h, h_bf);
        // logits = h_new @ fc_w^T + fc_b   (M=64, N=16000, K=1024)
        dec_gemm_bf16<<<VOCAB / 64, 128, 0, stream>>>(
            h_bf, fc_w_bf, fc_b, logits, VOCAB, HID);
        dec_logsoftmax<<<BATCH, 256, 0, stream>>>(
            logits, out + (size_t)t * VOCAB, inp);
    }

    // decoder_hidden appended after decoder_outputs
    dec_copy_hidden<<<(BATCH * HID + 255) / 256, 256, 0, stream>>>(
        h, out + (size_t)BATCH * T_LEN * VOCAB);
}
